// MultiHeadAttention_60576218742875
// MI455X (gfx1250) — matrix operand order repaired and checked by reading, not compiled, not run on hardware
//
#include <hip/hip_runtime.h>
#include <hip/hip_bf16.h>

typedef _Float16 f16;
typedef __attribute__((ext_vector_type(16))) _Float16 v16h;
typedef __attribute__((ext_vector_type(8)))  _Float16 h8;
typedef __attribute__((ext_vector_type(4)))  _Float16 h4;
typedef __attribute__((ext_vector_type(8)))  float    v8f;
typedef __attribute__((ext_vector_type(4)))  unsigned int u32x4;
typedef __attribute__((ext_vector_type(8)))  int i32x8;
typedef __attribute__((ext_vector_type(4)))  int i32x4;

#if __has_builtin(__builtin_amdgcn_tensor_load_to_lds)
#define HAVE_TDM 1
#else
#define HAVE_TDM 0
#endif

// ---------------------------------------------------------------------------
// WMMA wrapper: D = A(16x32 f16) * B(32x16 f16) + C(16x16 f32)
// ---------------------------------------------------------------------------
__device__ __forceinline__ v8f wmma(v16h a, v16h b, v8f c) {
  return __builtin_amdgcn_wmma_f32_16x16x32_f16(
      false, a, false, b, (short)0, c, false, false);
}

// A-fragment (16x32 f16): lane row m=lane&15, half=lane>>4.
// VGPR v holds pair k = 16*(v>>2) + 8*half + 2*(v&3)  =>
// elements [0..7] = row[8*half .. +7], [8..15] = row[16+8*half .. +7].
__device__ __forceinline__ v16h load_a_f16(const f16* p, int ld) {
  const int lane = threadIdx.x & 31;
  const f16* row = p + (size_t)(lane & 15) * ld + ((lane >> 4) << 3);
  h8 lo = *(const h8*)row;
  h8 hi = *(const h8*)(row + 16);
  return __builtin_shufflevector(lo, hi, 0, 1, 2, 3, 4, 5, 6, 7,
                                 8, 9, 10, 11, 12, 13, 14, 15);
}

// B-fragment (32x16 f16): lane col n=lane&15, khalf=lane>>4; k pair = 16*(v>>2)+8*kh+2*(v&3).
// Elements [0..7] = col[8*kh .. +7], [8..15] = col[16+8*kh .. +7].  Element (k,n) at p[n*ldn + k].
__device__ __forceinline__ v16h load_b_f16(const f16* p, int ldn) {
  const int lane = threadIdx.x & 31;
  const f16* col = p + (size_t)(lane & 15) * ldn + ((lane >> 4) << 3);
  h8 lo = *(const h8*)col;
  h8 hi = *(const h8*)(col + 16);
  return __builtin_shufflevector(lo, hi, 0, 1, 2, 3, 4, 5, 6, 7,
                                 8, 9, 10, 11, 12, 13, 14, 15);
}

#if HAVE_TDM
// ---------------------------------------------------------------------------
// Tensor Data Mover: 2D tile (rows x cols f16 elements) global -> LDS.
// D# per CDNA5 ISA 8.3/8.4.
// ---------------------------------------------------------------------------
__device__ __forceinline__ void tdm_load_tile_2d(unsigned lds_off, const void* gaddr,
                                                 int rows, int cols, int stride) {
  unsigned long long ga = (unsigned long long)gaddr;
  u32x4 g0;
  g0[0] = 1u;                                                  // count = 1
  g0[1] = lds_off;                                             // lds_addr (bytes)
  g0[2] = (unsigned)(ga & 0xffffffffu);                        // global_addr lo
  g0[3] = (unsigned)((ga >> 32) & 0x01ffffffu) | (2u << 30);   // addr hi | type=2
  i32x8 g1;
  g1[0] = 1 << 16;                                             // data_size = 1 (2B)
  g1[1] = (cols & 0xffff) << 16;                               // tensor_dim0 lo16
  g1[2] = ((cols >> 16) & 0xffff) | ((rows & 0xffff) << 16);   // dim0 hi | tensor_dim1 lo
  g1[3] = (cols & 0xffff) << 16;                               // tile_dim0
  g1[4] = rows & 0xffff;                                       // tile_dim1 (tile_dim2=0)
  g1[5] = stride;                                              // tensor_dim0_stride lo32
  g1[6] = 0;
  g1[7] = 0;
  i32x4 z4 = {0, 0, 0, 0};
#if defined(__clang_major__) && (__clang_major__ >= 23)
  i32x8 z8 = {0, 0, 0, 0, 0, 0, 0, 0};
  __builtin_amdgcn_tensor_load_to_lds(g0, g1, z4, z4, z8, 0);
#else
  __builtin_amdgcn_tensor_load_to_lds(g0, g1, z4, z4, 0);
#endif
}
#endif

// ---------------------------------------------------------------------------
// Kernel 1: positional tables.  Ak[r][e] (2048x64) row-major, AvT[e][r] (64x2048).
// ---------------------------------------------------------------------------
__global__ void k_tables(const float* __restrict__ Wak, const float* __restrict__ bak,
                         const float* __restrict__ Wav, const float* __restrict__ bav,
                         f16* __restrict__ Ak, f16* __restrict__ AvT) {
  int id = blockIdx.x * blockDim.x + threadIdx.x;   // 0 .. 262143
  int table = id >> 17;
  int idx = id & 131071;
  int r = idx >> 6, e = idx & 63;
  float arg = (1.5708f / 1024.0f) * (float)(r - 1023);
  float t0 = sinf(arg), t1 = cosf(arg);
  if (table == 0) {
    Ak[idx] = (_Float16)(t0 * Wak[e] + t1 * Wak[64 + e] + bak[e]);
  } else {
    AvT[(size_t)e * 2048 + r] = (_Float16)(t0 * Wav[e] + t1 * Wav[64 + e] + bav[e]);
  }
}

// ---------------------------------------------------------------------------
// Kernel 1b: f32 -> f16 cast (4 elements / thread).
// ---------------------------------------------------------------------------
__global__ void k_cast(const float* __restrict__ in, f16* __restrict__ out, int n4) {
  int id = blockIdx.x * blockDim.x + threadIdx.x;
  if (id >= n4) return;
  float4 v = ((const float4*)in)[id];
  h4 o;
  o[0] = (_Float16)v.x; o[1] = (_Float16)v.y;
  o[2] = (_Float16)v.z; o[3] = (_Float16)v.w;
  ((h4*)out)[id] = o;
}

// ---------------------------------------------------------------------------
// Kernel 1c: transpose + cast:  in f32 [H][R][C] -> out f16 [H][C][R].
// ---------------------------------------------------------------------------
__global__ void k_tcast(const float* __restrict__ in, f16* __restrict__ out,
                        int R, int C, int total) {
  int id = blockIdx.x * blockDim.x + threadIdx.x;
  if (id >= total) return;
  int rc = R * C;
  int hh = id / rc, rem = id - hh * rc;
  int cc = rem / R, rr = rem - cc * R;
  out[id] = (_Float16)in[(size_t)hh * rc + (size_t)rr * C + cc];
}

// ---------------------------------------------------------------------------
// Kernel 2: per-head projection GEMM, all-f16 operands.
// Xh (4096x1024 f16) @ WT (f16 [H][64 e][1024 d]) + bias -> f16 out.
// transposed==0: out[((b*16+h)*1024+l)*64+e]   (Qh/Kh)
// transposed==1: out[((b*16+h)*64+e)*1024+l]   (VhT)
// ---------------------------------------------------------------------------
__global__ __launch_bounds__(256) void k_proj(const f16* __restrict__ X,
                                              const f16* __restrict__ WT,
                                              const float* __restrict__ bias,
                                              f16* __restrict__ out, int transposed) {
  const int lane = threadIdx.x & 31;
  const int wave = blockIdx.x * 8 + (threadIdx.x >> 5);  // 0..16383
  const int m0 = (wave >> 6) << 4;
  const int n0 = (wave & 63) << 4;
  const int h = n0 >> 6, e0 = n0 & 63;

  v8f c = {};
#pragma unroll 4
  for (int k0 = 0; k0 < 1024; k0 += 32) {
    c = wmma(load_a_f16(X + (size_t)m0 * 1024 + k0, 1024),
             load_b_f16(WT + (size_t)h * 65536 + (size_t)e0 * 1024 + k0, 1024), c);
  }
  const int col = lane & 15, half = lane >> 4;
  const float bn = bias[h * 64 + e0 + col];
#pragma unroll
  for (int v = 0; v < 8; ++v) {
    int row = v + (half << 3);
    int mg = m0 + row;
    int bb = mg >> 10, l = mg & 1023;
    int e = e0 + col;
    size_t idx = transposed
        ? (((size_t)(bb * 16 + h) * 64 + e) * 1024 + l)
        : (((size_t)(bb * 16 + h) * 1024 + l) * 64 + e);
    out[idx] = (_Float16)(c[v] + bn);
  }
}

// ---------------------------------------------------------------------------
// Kernel 3: fused relative attention.  One wave = one (b,h, 16-row i-block);
// the 8 waves of a block share one (b,h) and cooperatively stage K/V tiles
// to LDS through the Tensor Data Mover (wave 0 issues DMA, split barriers
// publish).  j processed in steps of 32; online softmax; scores scaled 1/d.
// ---------------------------------------------------------------------------
__global__ __launch_bounds__(256) void k_attn(const f16* __restrict__ Qh,
                                              const f16* __restrict__ Kh,
                                              const f16* __restrict__ VhT,
                                              const f16* __restrict__ Ak,
                                              const f16* __restrict__ AvT,
                                              f16* __restrict__ AttOut) {
  __shared__ __align__(16) _Float16 ldsP[8][16][32];   // per-wave P tile (16x32)
#if HAVE_TDM
  __shared__ __align__(16) _Float16 ldsK[32][64];      // block-shared K tile
  __shared__ __align__(16) _Float16 ldsV[64][32];      // block-shared V tile (e-major)
#endif
  const int lane = threadIdx.x & 31;
  const int w = threadIdx.x >> 5;
  const int wave = blockIdx.x * 8 + w;       // 0..4095
  const int bh = wave >> 6;                  // 0..63 (identical for all 8 waves)
  const int i0 = (wave & 63) << 4;
  const int b = bh >> 4, h = bh & 15;

  const f16* Qbh = Qh + (size_t)bh * 1024 * 64;
  const f16* Kbh = Kh + (size_t)bh * 1024 * 64;
  const f16* Vbh = VhT + (size_t)bh * 64 * 1024;
#if HAVE_TDM
  const unsigned ldsKoff = (unsigned)(unsigned long long)&ldsK[0][0];
  const unsigned ldsVoff = (unsigned)(unsigned long long)&ldsV[0][0];
#endif

  const v16h aQ0 = load_a_f16(Qbh + i0 * 64, 64);       // K = 0..31
  const v16h aQ1 = load_a_f16(Qbh + i0 * 64 + 32, 64);  // K = 32..63

  float mrow[8], lrow[8];
  v8f acc[4];
#pragma unroll
  for (int v = 0; v < 8; ++v) { mrow[v] = -3.0e30f; lrow[v] = 0.0f; }
#pragma unroll
  for (int t = 0; t < 4; ++t) acc[t] = {};

  const int m = lane & 15, half = lane >> 4;
  const float inv_d = 1.0f / 64.0f;

  for (int j0 = 0; j0 < 1024; j0 += 32) {
    const int r0 = j0 - i0 + 1008;           // in [0, 2000]; band r0..r0+46 used
    __builtin_prefetch(Ak + (size_t)(r0 + 48) * 64, 0, 0);

#if HAVE_TDM
    // Drain reads of the previous tile, stage K (32x64) and V (64x32) via TDM.
    asm volatile("s_wait_dscnt 0" ::: "memory");
    __syncthreads();
    if (w == 0) {
      tdm_load_tile_2d(ldsKoff, Kbh + (size_t)j0 * 64, 32, 64, 64);
      tdm_load_tile_2d(ldsVoff, Vbh + j0, 64, 32, 1024);
      __builtin_amdgcn_s_wait_tensorcnt(0);
    }
    __syncthreads();
#endif

    // ---- base scores: Q(16x64) @ K(j0:j0+32)^T -> two 16x16 tiles ----
    v8f cS0 = {}, cS1 = {};
#if HAVE_TDM
    cS0 = wmma(aQ0, load_b_f16(&ldsK[0][0], 64), cS0);
    cS0 = wmma(aQ1, load_b_f16(&ldsK[0][32], 64), cS0);
    cS1 = wmma(aQ0, load_b_f16(&ldsK[16][0], 64), cS1);
    cS1 = wmma(aQ1, load_b_f16(&ldsK[16][32], 64), cS1);
#else
    cS0 = wmma(aQ0, load_b_f16(Kbh + (size_t)j0 * 64, 64), cS0);
    cS0 = wmma(aQ1, load_b_f16(Kbh + (size_t)j0 * 64 + 32, 64), cS0);
    cS1 = wmma(aQ0, load_b_f16(Kbh + (size_t)(j0 + 16) * 64, 64), cS1);
    cS1 = wmma(aQ1, load_b_f16(Kbh + (size_t)(j0 + 16) * 64 + 32, 64), cS1);
#endif

    // ---- rel scores: Rt = Q @ Ak[r0:r0+48]^T -> three 16x16 tiles ----
    v8f cR0 = {}, cR1 = {}, cR2 = {};
    cR0 = wmma(aQ0, load_b_f16(Ak + (size_t)r0 * 64, 64), cR0);
    cR0 = wmma(aQ1, load_b_f16(Ak + (size_t)r0 * 64 + 32, 64), cR0);
    cR1 = wmma(aQ0, load_b_f16(Ak + (size_t)(r0 + 16) * 64, 64), cR1);
    cR1 = wmma(aQ1, load_b_f16(Ak + (size_t)(r0 + 16) * 64 + 32, 64), cR1);
    cR2 = wmma(aQ0, load_b_f16(Ak + (size_t)(r0 + 32) * 64, 64), cR2);
    cR2 = wmma(aQ1, load_b_f16(Ak + (size_t)(r0 + 32) * 64 + 32, 64), cR2);

    // ---- skew gather + online softmax over 32 columns ----
    // Lane holds cols m (tile cS0) and m+16 (tile cS1) of row = v + 8*half.
    // s_rel[i,j'] = Rt[i, j'-i+15]; cols m / m+16 share a source lane.
#pragma unroll
    for (int v = 0; v < 8; ++v) {
      int row = v + (half << 3);
      int c0 = m - row + 15;                 // in [0,30]; c1 = c0+16 in [16,46]
      int src = (half << 4) | (c0 & 15);
      float g0 = __shfl(cR0[v], src, 32);
      float g1 = __shfl(cR1[v], src, 32);
      float g2 = __shfl(cR2[v], src, 32);
      float rel0 = (c0 < 16) ? g0 : g1;
      float rel1 = (c0 < 16) ? g1 : g2;
      float s0 = (cS0[v] + rel0) * inv_d;
      float s1 = (cS1[v] + rel1) * inv_d;

      float mt = fmaxf(s0, s1);
      mt = fmaxf(mt, __shfl_xor(mt, 1, 16));
      mt = fmaxf(mt, __shfl_xor(mt, 2, 16));
      mt = fmaxf(mt, __shfl_xor(mt, 4, 16));
      mt = fmaxf(mt, __shfl_xor(mt, 8, 16));
      float mn = fmaxf(mrow[v], mt);
      float corr = __expf(mrow[v] - mn);
      float p0 = __expf(s0 - mn);
      float p1 = __expf(s1 - mn);
      float ps = p0 + p1;
      ps += __shfl_xor(ps, 1, 16);
      ps += __shfl_xor(ps, 2, 16);
      ps += __shfl_xor(ps, 4, 16);
      ps += __shfl_xor(ps, 8, 16);
      lrow[v] = lrow[v] * corr + ps;
      mrow[v] = mn;
#pragma unroll
      for (int t = 0; t < 4; ++t) acc[t][v] *= corr;
      ldsP[w][row][m]      = (_Float16)p0;
      ldsP[w][row][m + 16] = (_Float16)p1;
    }

    // ---- A fragments from the wave-private P tile ----
    // aP: standard 16x32 fragment (vectorized LDS reads, full K).
    v16h aP = load_a_f16(&ldsP[w][0][0], 32);
    // aPl0/aPl1: skewed Pl[i,c] = P[i, c+i-15], c in [0,64) (valid c <= 46).
    v16h aPl0, aPl1;
    const _Float16* pr = &ldsP[w][m][0];
#pragma unroll
    for (int v = 0; v < 8; ++v) {
      int k = ((v >> 2) << 4) + (half << 3) + ((v & 3) << 1);
      int ja = k + m - 15;                   // c = k
      aPl0[2 * v]     = (ja >= 0 && ja < 32) ? pr[ja] : (_Float16)0.0f;
      ++ja;
      aPl0[2 * v + 1] = (ja >= 0 && ja < 32) ? pr[ja] : (_Float16)0.0f;
      int jb = k + m + 17;                   // c = k + 32  (jb >= 17)
      aPl1[2 * v]     = (jb < 32) ? pr[jb] : (_Float16)0.0f;
      ++jb;
      aPl1[2 * v + 1] = (jb < 32) ? pr[jb] : (_Float16)0.0f;
    }

    // ---- acc += P @ V  +  Pl @ Av[r0:r0+64]  (4 e-tiles of 16) ----
#pragma unroll
    for (int t = 0; t < 4; ++t) {
#if HAVE_TDM
      acc[t] = wmma(aP, load_b_f16(&ldsV[t * 16][0], 32), acc[t]);
#else
      acc[t] = wmma(aP, load_b_f16(Vbh + (size_t)t * 16 * 1024 + j0, 1024), acc[t]);
#endif
      acc[t] = wmma(aPl0, load_b_f16(AvT + (size_t)t * 16 * 2048 + r0, 2048), acc[t]);
      acc[t] = wmma(aPl1, load_b_f16(AvT + (size_t)t * 16 * 2048 + r0 + 32, 2048), acc[t]);
    }
  }

  // ---- normalize and store (concat-heads layout [b][l][h*64+e]) ----
#pragma unroll
  for (int t = 0; t < 4; ++t) {
#pragma unroll
    for (int v = 0; v < 8; ++v) {
      int row = v + (half << 3);
      int i = i0 + row;
      int e = t * 16 + m;
      float o = acc[t][v] / lrow[v];
      AttOut[((size_t)(b * 1024 + i)) * 1024 + h * 64 + e] = (_Float16)o;
    }
  }
}

// ---------------------------------------------------------------------------
// Kernel 4: output dense + mish, all-f16 operands.
// AttOut(4096x1024 f16) @ WdT(f16 [n][k]) + bd -> f32 out.
// ---------------------------------------------------------------------------
__global__ __launch_bounds__(256) void k_dense(const f16* __restrict__ A,
                                               const f16* __restrict__ WdT,
                                               const float* __restrict__ bd,
                                               float* __restrict__ out) {
  const int lane = threadIdx.x & 31;
  const int wave = blockIdx.x * 8 + (threadIdx.x >> 5);
  const int m0 = (wave >> 6) << 4;
  const int n0 = (wave & 63) << 4;

  v8f c = {};
#pragma unroll 4
  for (int k0 = 0; k0 < 1024; k0 += 32) {
    c = wmma(load_a_f16(A + (size_t)m0 * 1024 + k0, 1024),
             load_b_f16(WdT + (size_t)n0 * 1024 + k0, 1024), c);
  }
  const int col = lane & 15, half = lane >> 4;
  const float bn = bd[n0 + col];
#pragma unroll
  for (int v = 0; v < 8; ++v) {
    int row = v + (half << 3);
    float xv = c[v] + bn;
    float sp = (xv > 20.0f) ? xv : log1pf(__expf(xv));
    out[(size_t)(m0 + row) * 1024 + n0 + col] = xv * tanhf(sp);
  }
}

// ---------------------------------------------------------------------------
// Host-side launcher
// ---------------------------------------------------------------------------
extern "C" void kernel_launch(void* const* d_in, const int* in_sizes, int n_in,
                              void* d_out, int out_size, void* d_ws, size_t ws_size,
                              hipStream_t stream) {
  (void)in_sizes; (void)n_in; (void)out_size; (void)ws_size;
  const float* x   = (const float*)d_in[0];
  const float* kk  = (const float*)d_in[1];
  const float* q   = (const float*)d_in[2];
  const float* Wq  = (const float*)d_in[4];
  const float* bq  = (const float*)d_in[5];
  const float* Wk  = (const float*)d_in[6];
  const float* bk  = (const float*)d_in[7];
  const float* Wv  = (const float*)d_in[8];
  const float* bv  = (const float*)d_in[9];
  const float* Wak = (const float*)d_in[10];
  const float* bak = (const float*)d_in[11];
  const float* Wav = (const float*)d_in[12];
  const float* bav = (const float*)d_in[13];
  const float* Wd  = (const float*)d_in[14];
  const float* bd  = (const float*)d_in[15];

  f16* ws     = (f16*)d_ws;
  f16* Qh     = ws;                 // [64 bh][1024][64]   4,194,304
  f16* Kh     = Qh  + 4194304;      // [64 bh][1024][64]
  f16* VhT    = Kh  + 4194304;      // [64 bh][64][1024]
  f16* Ak     = VhT + 4194304;      // [2048][64]            131,072
  f16* AvT    = Ak  + 131072;       // [64][2048]            131,072
  f16* AttOut = AvT + 131072;       // [4][1024][1024]     4,194,304
  f16* Xh     = AttOut + 4194304;   // reusable f16 input   4,194,304
  f16* WT     = Xh + 4194304;       // reusable f16 weightT 1,048,576
  // total ~22.3M f16 = ~42.5 MB (+ slack for benign zero-weighted overreads)

  k_tables<<<1024, 256, 0, stream>>>(Wak, bak, Wav, bav, Ak, AvT);

  // Q projection
  k_cast<<<4096, 256, 0, stream>>>(q, Xh, 1048576);
  k_tcast<<<4096, 256, 0, stream>>>(Wq, WT, 1024, 64, 1048576);
  k_proj<<<2048, 256, 0, stream>>>(Xh, WT, bq, Qh, 0);
  // K projection
  k_cast<<<4096, 256, 0, stream>>>(kk, Xh, 1048576);
  k_tcast<<<4096, 256, 0, stream>>>(Wk, WT, 1024, 64, 1048576);
  k_proj<<<2048, 256, 0, stream>>>(Xh, WT, bk, Kh, 0);
  // V projection (transposed output)
  k_cast<<<4096, 256, 0, stream>>>(x, Xh, 1048576);
  k_tcast<<<4096, 256, 0, stream>>>(Wv, WT, 1024, 64, 1048576);
  k_proj<<<2048, 256, 0, stream>>>(Xh, WT, bv, VhT, 1);

  k_attn<<<512, 256, 0, stream>>>(Qh, Kh, VhT, Ak, AvT, AttOut);

  k_tcast<<<4096, 256, 0, stream>>>(Wd, WT, 1024, 1024, 1048576);
  k_dense<<<2048, 256, 0, stream>>>(AttOut, WT, bd, (float*)d_out);
}
